// ScaledDotProductAttention_86517821214099
// MI455X (gfx1250) — compile-verified, matrix operand order checked by reading, not run on hardware
//
#include <hip/hip_runtime.h>
#include <hip/hip_bf16.h>
#include <math.h>

typedef __attribute__((ext_vector_type(16))) _Float16 v16h;
typedef __attribute__((ext_vector_type(8)))  float    v8f;
typedef __attribute__((ext_vector_type(4)))  float    nf4;   // native vector for NT stores

namespace {
constexpr int NB   = 2;
constexpr int NH   = 16;
constexpr int SEQ  = 2048;
constexpr int HD   = 64;
constexpr int LDP  = SEQ + 4;          // padded LDS row stride (floats), 16B aligned
constexpr int NWAVES = 8;
constexpr int KSLICE = SEQ / NWAVES;   // 256 keys per wave
constexpr float INV_T   = 0.125f;      // 1 / sqrt(64)
constexpr float NEG_INF = -1e9f;
}

// Load a 16-element f16 A/B-tile vector for one lane from a row-major f32 row.
// CDNA5 16-bit 16x32 striping: lane holds halves at k = kbase+khalf+[0..7] and
// kbase+16+khalf+[0..7]. kbase,khalf multiples of 8 -> float4 aligned.
__device__ inline v16h load_tile_f16(const float* rowp, int kbase, int khalf, float scale) {
  const float* p = rowp + kbase + khalf;
  float4 x0 = *(const float4*)(p + 0);
  float4 x1 = *(const float4*)(p + 4);
  float4 x2 = *(const float4*)(p + 16);
  float4 x3 = *(const float4*)(p + 20);
  v16h a;
  a[0]  = (_Float16)(x0.x * scale); a[1]  = (_Float16)(x0.y * scale);
  a[2]  = (_Float16)(x0.z * scale); a[3]  = (_Float16)(x0.w * scale);
  a[4]  = (_Float16)(x1.x * scale); a[5]  = (_Float16)(x1.y * scale);
  a[6]  = (_Float16)(x1.z * scale); a[7]  = (_Float16)(x1.w * scale);
  a[8]  = (_Float16)(x2.x * scale); a[9]  = (_Float16)(x2.y * scale);
  a[10] = (_Float16)(x2.z * scale); a[11] = (_Float16)(x2.w * scale);
  a[12] = (_Float16)(x3.x * scale); a[13] = (_Float16)(x3.y * scale);
  a[14] = (_Float16)(x3.z * scale); a[15] = (_Float16)(x3.w * scale);
  return a;
}

// B-tile from V: column d (strided by HD), rows kbase+khalf+[0..7] and +16.
__device__ inline v16h load_vtile_f16(const float* __restrict__ vb, int kbase, int khalf, int d) {
  v16h b;
#pragma unroll
  for (int j = 0; j < 8; ++j)
    b[j] = (_Float16)vb[(size_t)(kbase + khalf + j) * HD + d];
#pragma unroll
  for (int j = 0; j < 8; ++j)
    b[8 + j] = (_Float16)vb[(size_t)(kbase + 16 + khalf + j) * HD + d];
  return b;
}

__global__ __launch_bounds__(256)
void sdpa_fused_kernel(const float* __restrict__ q, const float* __restrict__ k,
                       const float* __restrict__ v, const int* __restrict__ mask,
                       float* __restrict__ out, float* __restrict__ attn) {
  __shared__ float sc[16 * LDP];   // scores / P, reused for partial-O reduction
  __shared__ float red[16 * 16];   // softmax row reductions

  const int tid   = threadIdx.x;
  const int lane  = tid & 31;
  const int wave  = tid >> 5;
  const int bh    = blockIdx.y;            // 0..31
  const int b     = bh / NH;
  const int qbase = blockIdx.x * 16;       // 0..2032

  const float* qb = q + (size_t)bh * SEQ * HD + (size_t)qbase * HD;
  const float* kb = k + (size_t)bh * SEQ * HD;
  const float* vb = v + (size_t)bh * SEQ * HD;
  const int*   mb = mask + (size_t)b * SEQ * SEQ + (size_t)qbase * SEQ;

  const int arow  = lane & 15;           // A row / B,C,D column within tile
  const int khalf = (lane >> 4) * 8;     // K-half offset; also C/D M-offset

  // ---------------- Stage 1: scores = (Q/T) @ K^T, masked, into LDS ----------------
  const v16h aq0 = load_tile_f16(qb + (size_t)arow * HD, 0,  khalf, INV_T);
  const v16h aq1 = load_tile_f16(qb + (size_t)arow * HD, 32, khalf, INV_T);

  for (int chunk = 0; chunk < KSLICE / 16; ++chunk) {
    const int cbase = wave * KSLICE + chunk * 16;
    const int col   = cbase + arow;                    // key index for this lane
    // prefetch next chunk's K row toward the WGP while this chunk computes
    if (chunk + 1 < KSLICE / 16)
      __builtin_prefetch(kb + (size_t)(col + 16) * HD, 0, 3);

    v16h bk0 = load_tile_f16(kb + (size_t)col * HD, 0,  khalf, 1.0f);
    v16h bk1 = load_tile_f16(kb + (size_t)col * HD, 32, khalf, 1.0f);
    v8f c = {};
    c = __builtin_amdgcn_wmma_f32_16x16x32_f16(false, aq0, false, bk0, (short)0, c, false, false);
    c = __builtin_amdgcn_wmma_f32_16x16x32_f16(false, aq1, false, bk1, (short)0, c, false, false);
#pragma unroll
    for (int r = 0; r < 8; ++r) {
      const int M = r + khalf;                         // C/D row layout
      const int m = mb[(size_t)M * SEQ + col];
      sc[M * LDP + col] = (m == 0) ? NEG_INF : c[r];
    }
  }
  __syncthreads();

  // ---------------- Stage 2: row softmax + stream attn to HBM (non-temporal) ----------------
  {
    const int row = tid >> 4;                // 16 rows, 16 threads each
    const int seg = tid & 15;                // contiguous 128-float segment
    float* rp = &sc[row * LDP + seg * 128];

    float mx = -INFINITY;
    for (int i = 0; i < 128; i += 4) {
      float4 x = *(const float4*)(rp + i);
      mx = fmaxf(mx, fmaxf(fmaxf(x.x, x.y), fmaxf(x.z, x.w)));
    }
    red[row * 16 + seg] = mx;
    __syncthreads();
    float rowmax = red[row * 16];
#pragma unroll
    for (int j = 1; j < 16; ++j) rowmax = fmaxf(rowmax, red[row * 16 + j]);
    __syncthreads();

    float s = 0.0f;
    for (int i = 0; i < 128; i += 4) {
      float4 x = *(const float4*)(rp + i);
      x.x = __expf(x.x - rowmax); x.y = __expf(x.y - rowmax);
      x.z = __expf(x.z - rowmax); x.w = __expf(x.w - rowmax);
      s += x.x + x.y + x.z + x.w;
      *(float4*)(rp + i) = x;
    }
    red[row * 16 + seg] = s;
    __syncthreads();
    float rowsum = 0.0f;
#pragma unroll
    for (int j = 0; j < 16; ++j) rowsum += red[row * 16 + j];
    const float inv = 1.0f / rowsum;

    float* attnp = attn + ((size_t)bh * SEQ + qbase + row) * SEQ + seg * 128;
    for (int i = 0; i < 128; i += 4) {
      float4 x = *(const float4*)(rp + i);
      x.x *= inv; x.y *= inv; x.z *= inv; x.w *= inv;
      *(float4*)(rp + i) = x;                              // normalized P for stage 3
      nf4 nx = {x.x, x.y, x.z, x.w};
      __builtin_nontemporal_store(nx, (nf4*)(attnp + i));  // streamed attn output (TH=NT)
    }
  }
  __syncthreads();

  // ---------------- Stage 3: out = P @ V ----------------
  v8f acc0 = {}, acc1 = {}, acc2 = {}, acc3 = {};
  for (int chunk = 0; chunk < KSLICE / 32; ++chunk) {
    const int ck = wave * KSLICE + chunk * 32;
    v16h ap = load_tile_f16(&sc[arow * LDP], ck, khalf, 1.0f);   // P tile (16x32)
    v16h b0 = load_vtile_f16(vb, ck, khalf, 0  + arow);
    acc0 = __builtin_amdgcn_wmma_f32_16x16x32_f16(false, ap, false, b0, (short)0, acc0, false, false);
    v16h b1 = load_vtile_f16(vb, ck, khalf, 16 + arow);
    acc1 = __builtin_amdgcn_wmma_f32_16x16x32_f16(false, ap, false, b1, (short)0, acc1, false, false);
    v16h b2 = load_vtile_f16(vb, ck, khalf, 32 + arow);
    acc2 = __builtin_amdgcn_wmma_f32_16x16x32_f16(false, ap, false, b2, (short)0, acc2, false, false);
    v16h b3 = load_vtile_f16(vb, ck, khalf, 48 + arow);
    acc3 = __builtin_amdgcn_wmma_f32_16x16x32_f16(false, ap, false, b3, (short)0, acc3, false, false);
  }
  __syncthreads();   // everyone done reading P from sc

  // spill per-wave partial 16x64 O into (reused) score LDS, flat layout
  float* part = sc;
#pragma unroll
  for (int r = 0; r < 8; ++r) {
    const int M = r + khalf;
    part[wave * 1024 + M * HD + 0  + arow] = acc0[r];
    part[wave * 1024 + M * HD + 16 + arow] = acc1[r];
    part[wave * 1024 + M * HD + 32 + arow] = acc2[r];
    part[wave * 1024 + M * HD + 48 + arow] = acc3[r];
  }
  __syncthreads();

  // cross-wave reduction: 1024 outputs, 4 per thread (float4, NT store)
  {
    const int e = tid * 4;
    nf4 sum = {0.f, 0.f, 0.f, 0.f};
#pragma unroll
    for (int w = 0; w < NWAVES; ++w) {
      float4 x = *(const float4*)&part[w * 1024 + e];
      sum.x += x.x; sum.y += x.y; sum.z += x.z; sum.w += x.w;
    }
    const int M = e >> 6;        // /64
    const int d = e & 63;
    __builtin_nontemporal_store(sum, (nf4*)(out + ((size_t)bh * SEQ + qbase + M) * HD + d));
  }
}

extern "C" void kernel_launch(void* const* d_in, const int* in_sizes, int n_in,
                              void* d_out, int out_size, void* d_ws, size_t ws_size,
                              hipStream_t stream) {
  const float* q    = (const float*)d_in[0];
  const float* k    = (const float*)d_in[1];
  const float* v    = (const float*)d_in[2];
  const int*   mask = (const int*)d_in[3];

  float* out  = (float*)d_out;                         // [B,H,S,D] first
  float* attn = out + (size_t)NB * NH * SEQ * HD;      // then [B,H,S,S]

  dim3 grid(SEQ / 16, NB * NH);   // 128 x 32 workgroups
  dim3 block(256);                // 8 waves (wave32)
  sdpa_fused_kernel<<<grid, block, 0, stream>>>(q, k, v, mask, out, attn);
}
